// GIN_51410758533587
// MI455X (gfx1250) — compile-verified, matrix operand order checked by reading, not run on hardware
//
#include <hip/hip_runtime.h>
#include <hip/hip_bf16.h>

// ---------- constants from the reference ----------
#define GN  50000      // nodes
#define GE  600000     // edges
#define GF  128        // in features
#define GH  128        // hidden
#define GG  512        // graphs
#define GC  64         // classes
#define BN_EPS 1e-5f

typedef __attribute__((ext_vector_type(16))) __bf16 v16bf;
typedef __attribute__((ext_vector_type(8)))  __bf16 v8bf;
typedef __attribute__((ext_vector_type(8)))  float  v8f;

// =====================================================================
// Register-blocked bf16 WMMA GEMM.
//   out[M,Nd] = epilogue( A[M,K] @ W[Nd,K]^T )
// One wave32 computes a 16-row x (NT*16)-col strip: NT v8f accumulators,
// A fragment loaded once per 32-wide K chunk and reused by NT WMMAs.
//   AF32:  A is float32 (optionally plus Aadd), converted to bf16 in-regs
//          (fuses the "x + agg -> bf16" pass into the GEMM).
//   POOL:  epilogue also does pooled[batch[row]*3H + poolOff + col] += val
//          (fuses global_add_pool into the layer-output GEMM).
// epilogue: val = acc*scale[col] + shift[col]; optional ReLU.
// Requires M%16==0, K%32==0, (Nd/16)%NT==0 (true for all shapes here).
// =====================================================================
template<int NT, bool AF32, bool POOL>
__global__ void wmma_gemm_rb(const void* __restrict__ Aptr,
                             const float* __restrict__ Aadd,
                             const __bf16* __restrict__ W,
                             const float* __restrict__ scale,
                             const float* __restrict__ shift,
                             float* __restrict__ outF,
                             __bf16* __restrict__ outB,
                             const int* __restrict__ batch,
                             float* __restrict__ pooled, int poolOff,
                             int M, int Nd, int K, int relu)
{
    const int wave = (int)((blockIdx.x * (unsigned)blockDim.x + threadIdx.x) >> 5);
    const int lane = threadIdx.x & 31;
    const int colGroups = (Nd >> 4) / NT;
    const int total = (M >> 4) * colGroups;
    if (wave >= total) return;              // whole waves exit -> EXEC all-1s inside

    const int rt = wave / colGroups;        // 16-row tile index
    const int cg = wave % colGroups;        // column group (NT tiles wide)
    const int half = lane >> 4;             // 0: K{0..7,16..23}  1: K{8..15,24..31}
    const int l16  = lane & 15;
    const int row  = (rt << 4) + l16;       // A fragment row for this lane

    // A-side per-lane base (element offset of first 8-wide chunk)
    const size_t aoff = (size_t)row * K + half * 8;
    const float*  af32 = (const float*)Aptr + aoff;
    const float*  ag32 = Aadd ? (Aadd + aoff) : nullptr;
    const __bf16* abf  = (const __bf16*)Aptr + aoff;

    // B-side: column of tile ct for this lane = (cg*NT+ct)*16 + l16,
    // fragment base = W + col*K + half*8; consecutive tiles stride 16*K.
    const __bf16* __restrict__ bbase =
        W + ((size_t)((cg * NT) << 4) + l16) * K + half * 8;

    v8f acc[NT];
#pragma unroll
    for (int ct = 0; ct < NT; ++ct) acc[ct] = (v8f){};

    for (int kc = 0; kc < K; kc += 32) {
        // ---- A fragment (lane holds K chunks {half*8..+8} and {16+half*8..+8}) ----
        v16bf afrag;
        if (AF32) {
            v8f va = *(const v8f*)(af32 + kc);
            v8f vb = *(const v8f*)(af32 + kc + 16);
            if (ag32) {
                va += *(const v8f*)(ag32 + kc);
                vb += *(const v8f*)(ag32 + kc + 16);
            }
            v8bf c0, c1;
#pragma unroll
            for (int i = 0; i < 8; ++i) { c0[i] = (__bf16)va[i]; c1[i] = (__bf16)vb[i]; }
            afrag = __builtin_shufflevector(c0, c1, 0,1,2,3,4,5,6,7,8,9,10,11,12,13,14,15);
        } else {
            v8bf c0 = *(const v8bf*)(abf + kc);
            v8bf c1 = *(const v8bf*)(abf + kc + 16);
            afrag = __builtin_shufflevector(c0, c1, 0,1,2,3,4,5,6,7,8,9,10,11,12,13,14,15);
        }
        // ---- NT B fragments, NT WMMAs against the same A fragment ----
#pragma unroll
        for (int ct = 0; ct < NT; ++ct) {
            const __bf16* bp = bbase + (size_t)ct * 16 * K + kc;
            v8bf b0 = *(const v8bf*)(bp);
            v8bf b1 = *(const v8bf*)(bp + 16);
            v16bf bfrag = __builtin_shufflevector(b0, b1, 0,1,2,3,4,5,6,7,8,9,10,11,12,13,14,15);
            acc[ct] = __builtin_amdgcn_wmma_f32_16x16x32_bf16(
                false, afrag, false, bfrag, (short)0, acc[ct], false, false);
        }
    }

    // ---- epilogue ----
    const int rbase = (rt << 4) + half * 8; // C/D: lanes<16 rows 0..7, lanes>=16 rows 8..15
    int gb[8];
    if (POOL) {
#pragma unroll
        for (int i = 0; i < 8; ++i) gb[i] = batch[rbase + i];
    }
#pragma unroll
    for (int ct = 0; ct < NT; ++ct) {
        const int col = (((cg * NT) + ct) << 4) + l16;
        const float s = scale[col];
        const float t = shift[col];
#pragma unroll
        for (int i = 0; i < 8; ++i) {
            float v = acc[ct][i] * s + t;
            if (relu) v = fmaxf(v, 0.0f);
            const size_t idx = (size_t)(rbase + i) * Nd + col;
            if (outF) outF[idx] = v;
            if (outB) outB[idx] = (__bf16)v;
            if (POOL) atomicAdd(pooled + (size_t)gb[i] * (3 * GH) + poolOff + col, v);
        }
    }
}

// ---------- edge scatter-add: agg[dst] += x[src], one thread per (edge, float4) ----------
__global__ void scatter_add_edges(const float* __restrict__ x,
                                  const int* __restrict__ src,
                                  const int* __restrict__ dst,
                                  float* __restrict__ agg)
{
    const long long t = (long long)blockIdx.x * blockDim.x + threadIdx.x;
    if (t >= (long long)GE * 32) return;
    const int e = (int)(t >> 5);
    const int g = (int)(t & 31) << 2;
    const int sN = src[e];
    const int dN = dst[e];
    const float4 v = *(const float4*)(x + (size_t)sN * GF + g);
    float* p = agg + (size_t)dN * GF + g;
    atomicAdd(p + 0, v.x);
    atomicAdd(p + 1, v.y);
    atomicAdd(p + 2, v.z);
    atomicAdd(p + 3, v.w);
}

// ---------- f32 -> bf16 (weights) ----------
__global__ void cvt_bf16(const float* __restrict__ a, __bf16* __restrict__ o, int n)
{
    const int i = blockIdx.x * blockDim.x + threadIdx.x;
    if (i < n) o[i] = (__bf16)a[i];
}

// ---------- fold BatchNorm(eval)+bias into per-channel scale/shift ----------
__global__ void bn_fold(const float* __restrict__ bias, const float* __restrict__ g,
                        const float* __restrict__ be,   const float* __restrict__ m,
                        const float* __restrict__ v,
                        float* __restrict__ scale, float* __restrict__ shift, int n)
{
    const int i = blockIdx.x * blockDim.x + threadIdx.x;
    if (i >= n) return;
    const float s = g[i] * rsqrtf(v[i] + BN_EPS);
    scale[i] = s;
    shift[i] = (bias[i] - m[i]) * s + be[i];
}

__global__ void bias_fold(const float* __restrict__ b,
                          float* __restrict__ scale, float* __restrict__ shift, int n)
{
    const int i = blockIdx.x * blockDim.x + threadIdx.x;
    if (i >= n) return;
    scale[i] = 1.0f;
    shift[i] = b[i];
}

// =====================================================================
// Host driver
// =====================================================================
static inline size_t alignUp(size_t x, size_t a) { return (x + a - 1) / a * a; }

extern "C" void kernel_launch(void* const* d_in, const int* in_sizes, int n_in,
                              void* d_out, int out_size, void* d_ws, size_t ws_size,
                              hipStream_t stream)
{
    (void)in_sizes; (void)n_in; (void)out_size; (void)ws_size;

    // ----- inputs (setup_inputs dict order) -----
    const float* x     = (const float*)d_in[0];
    const int*   ei    = (const int*)d_in[1];   // [2, E]
    const int*   batch = (const int*)d_in[2];
    const int* src = ei;
    const int* dst = ei + GE;
    const float* wa[3], *ba[3], *bng[3], *bnb[3], *bnm[3], *bnv[3], *wb[3], *bb[3];
    for (int l = 0; l < 3; ++l) {
        const int o = 3 + 8 * l;
        wa[l]  = (const float*)d_in[o + 0];
        ba[l]  = (const float*)d_in[o + 1];
        bng[l] = (const float*)d_in[o + 2];
        bnb[l] = (const float*)d_in[o + 3];
        bnm[l] = (const float*)d_in[o + 4];
        bnv[l] = (const float*)d_in[o + 5];
        wb[l]  = (const float*)d_in[o + 6];
        bb[l]  = (const float*)d_in[o + 7];
    }
    const float* lin1_w = (const float*)d_in[27];
    const float* lin1_b = (const float*)d_in[28];
    const float* lin2_w = (const float*)d_in[29];
    const float* lin2_b = (const float*)d_in[30];
    float* out = (float*)d_out;

    // ----- workspace carve-out -----
    char* p = (char*)d_ws;
    auto alloc = [&](size_t bytes) -> void* {
        void* r = (void*)p;
        p += alignUp(bytes, 256);
        return r;
    };
    const size_t NF = (size_t)GN * GF;
    float*  agg    = (float*)alloc(NF * 4);
    float*  buf0   = (float*)alloc(NF * 4);   // h1, then h3
    float*  buf1   = (float*)alloc(NF * 4);   // h2
    __bf16* mB     = (__bf16*)alloc(NF * 2);  // GEMM1 output (bf16)
    float*  pooled = (float*)alloc((size_t)GG * 3 * GH * 4);
    __bf16* lin1actB = (__bf16*)alloc((size_t)GG * 3 * GH * 2);
    __bf16* waB[3], *wbB[3];
    for (int l = 0; l < 3; ++l) {
        waB[l] = (__bf16*)alloc((size_t)GH * GF * 2);
        wbB[l] = (__bf16*)alloc((size_t)GH * GH * 2);
    }
    __bf16* lin1wB = (__bf16*)alloc((size_t)3 * GH * 3 * GH * 2);
    __bf16* lin2wB = (__bf16*)alloc((size_t)GC * 3 * GH * 2);
    float *sc1[3], *sh1[3], *sc2[3], *sh2[3];
    for (int l = 0; l < 3; ++l) {
        sc1[l] = (float*)alloc(GH * 4); sh1[l] = (float*)alloc(GH * 4);
        sc2[l] = (float*)alloc(GH * 4); sh2[l] = (float*)alloc(GH * 4);
    }
    float* scL1 = (float*)alloc(3 * GH * 4); float* shL1 = (float*)alloc(3 * GH * 4);
    float* scL2 = (float*)alloc(GC * 4);     float* shL2 = (float*)alloc(GC * 4);

    const int TB = 256;                         // 8 waves / block
    auto blk = [](long long n, int t) { return (unsigned)((n + t - 1) / t); };
    auto gemmBlks = [](int M, int Nd, int NT) {
        const unsigned waves = (unsigned)((M / 16) * ((Nd / 16) / NT));
        return (waves + 7) / 8;
    };

    // ----- parameter prep (bf16 weights + folded epilogues) -----
    for (int l = 0; l < 3; ++l) {
        cvt_bf16<<<blk(GH * GF, TB), TB, 0, stream>>>(wa[l], waB[l], GH * GF);
        cvt_bf16<<<blk(GH * GH, TB), TB, 0, stream>>>(wb[l], wbB[l], GH * GH);
        bn_fold<<<1, GH, 0, stream>>>(ba[l], bng[l], bnb[l], bnm[l], bnv[l],
                                      sc1[l], sh1[l], GH);
        bias_fold<<<1, GH, 0, stream>>>(bb[l], sc2[l], sh2[l], GH);
    }
    cvt_bf16<<<blk(3 * GH * 3 * GH, TB), TB, 0, stream>>>(lin1_w, lin1wB, 3 * GH * 3 * GH);
    cvt_bf16<<<blk(GC * 3 * GH, TB), TB, 0, stream>>>(lin2_w, lin2wB, GC * 3 * GH);
    bias_fold<<<blk(3 * GH, TB), TB, 0, stream>>>(lin1_b, scL1, shL1, 3 * GH);
    bias_fold<<<1, GC, 0, stream>>>(lin2_b, scL2, shL2, GC);

    hipMemsetAsync(pooled, 0, (size_t)GG * 3 * GH * 4, stream);

    // ----- 3 GIN layers -----
    const float* inF[3]  = { x,    buf0, buf1 };
    float*       outL[3] = { buf0, buf1, buf0 };
    const unsigned gB = gemmBlks(GN, GH, 8);   // 3125 waves -> 391 blocks

    for (int l = 0; l < 3; ++l) {
        hipMemsetAsync(agg, 0, NF * 4, stream);
        scatter_add_edges<<<blk((long long)GE * 32, TB), TB, 0, stream>>>(inF[l], src, dst, agg);
        // Linear1 (+fused x+agg, bf16 cvt) + BN + ReLU -> bf16
        wmma_gemm_rb<8, true, false><<<gB, TB, 0, stream>>>(
            inF[l], agg, waB[l], sc1[l], sh1[l],
            nullptr, mB, nullptr, nullptr, 0, GN, GH, GF, 1);
        // Linear2 + bias + ReLU -> f32 layer output, fused global_add_pool
        wmma_gemm_rb<8, false, true><<<gB, TB, 0, stream>>>(
            mB, nullptr, wbB[l], sc2[l], sh2[l],
            outL[l], nullptr, batch, pooled, l * GH, GN, GH, GH, 1);
    }

    // ----- readout MLP on [G, 3H] -----
    // lin1: pooled (f32, fused cvt) @ lin1_w^T + b, ReLU -> bf16
    wmma_gemm_rb<8, true, false><<<gemmBlks(GG, 3 * GH, 8), TB, 0, stream>>>(
        pooled, nullptr, lin1wB, scL1, shL1,
        nullptr, lin1actB, nullptr, nullptr, 0, GG, 3 * GH, 3 * GH, 1);
    // lin2: -> d_out (f32), no ReLU
    wmma_gemm_rb<4, false, false><<<gemmBlks(GG, GC, 4), TB, 0, stream>>>(
        lin1actB, nullptr, lin2wB, scL2, shL2,
        out, nullptr, nullptr, nullptr, 0, GG, GC, 3 * GH, 0);
}
